// BidafAttn_68831145886537
// MI455X (gfx1250) — matrix operand order repaired and checked by reading, not compiled, not run on hardware
//
#include <hip/hip_runtime.h>
#include <hip/hip_bf16.h>
#include <math.h>

// Problem sizes (fixed by the reference).
#define BATCH 16
#define CLEN  4096
#define QLEN  512
#define HDIM  256

// TDM path: probe-verified 5-arg builtin on ROCm 7.2 (clang-22). clang-23+
// (amdgpu-toolchain) has a 6-arg variant we don't target -> cooperative copy.
#if defined(__clang_major__) && (__clang_major__ <= 22) && \
    __has_builtin(__builtin_amdgcn_tensor_load_to_lds) && \
    __has_builtin(__builtin_amdgcn_s_wait_tensorcnt)
#define TDM_PATH 1
#define NQBUF 2
#else
#define TDM_PATH 0
#define NQBUF 1
#endif

typedef __attribute__((ext_vector_type(16))) __bf16 v16bf;
typedef __attribute__((ext_vector_type(8)))  float  v8f;

union ABPack {
    v16bf v;
    uint4 u[2];
    unsigned int w[8];
};

__device__ __forceinline__ unsigned short f2bf(float f) {
    union { float f; unsigned int u; } x;
    x.f = f;
    unsigned int r = x.u + 0x7FFFu + ((x.u >> 16) & 1u);  // round-to-nearest-even
    return (unsigned short)(r >> 16);
}

#if TDM_PATH
// TDM descriptor per CDNA5 ISA (D# groups 0/1):
// 2-D tile: 64B rows x 256 rows, tensor row stride 1024B, data_size=4B,
// pad_enable: +16B LDS pad after every 64B (pad_interval=3 -> 16 DW,
// pad_amount=3 -> 4 DW) => 80B LDS row stride (bank-conflict-free b128 reads).
__device__ __forceinline__ void tdm_load_qtt(const unsigned short* gsrc,
                                             unsigned lds_addr) {
    typedef unsigned int v4u __attribute__((ext_vector_type(4)));
    typedef int v8i __attribute__((ext_vector_type(8)));
    typedef int v4i __attribute__((ext_vector_type(4)));
    const unsigned long long ga = (unsigned long long)(uintptr_t)gsrc;
    v4u g0;
    g0[0] = 1u;                                   // count=1 (valid user D#)
    g0[1] = lds_addr;                             // LDS byte address
    g0[2] = (unsigned)ga;                         // global_addr[31:0]
    g0[3] = (unsigned)((ga >> 32) & 0x01FFFFFFull) | (2u << 30);  // [56:32]|type=2
    const unsigned td0 = 256, td1 = 256;          // tensor dims (4B units / rows)
    const unsigned tile0 = 16, tile1 = 256;       // tile: 16x4B=64B rows, 256 rows
    const unsigned stride0 = 256;                 // 1024B row stride in 4B units
    v8i g1;
    g1[0] = (int)((2u << 16) | (1u << 20) | (3u << 22) | (3u << 25));
    g1[1] = (int)((td0 & 0xFFFFu) << 16);
    g1[2] = (int)(((td0 >> 16) & 0xFFFFu) | ((td1 & 0xFFFFu) << 16));
    g1[3] = (int)(((td1 >> 16) & 0xFFFFu) | ((tile0 & 0xFFFFu) << 16));
    g1[4] = (int)(tile1 & 0xFFFFu);               // tile_dim1; tile_dim2=0
    g1[5] = (int)stride0;                         // tensor_dim0_stride[31:0]
    g1[6] = 0;
    g1[7] = 0;
    v4i gz;
    gz[0] = gz[1] = gz[2] = gz[3] = 0;            // groups 2/3 unused (2-D)
    __builtin_amdgcn_tensor_load_to_lds(g0, g1, gz, gz, 0);
}
#endif

// ---------------------------------------------------------------------------
// Prologue: q_bf = bf16(q) (j x d); qT_bf = bf16(q)^T (d x j);
//           qw[row] = dot(q_row, w_q) + b_q
// ---------------------------------------------------------------------------
__global__ __launch_bounds__(256) void k_prep_q(
    const float* __restrict__ q, const float* __restrict__ w_q,
    const float* __restrict__ b_q,
    unsigned short* __restrict__ q_bf, unsigned short* __restrict__ qT_bf,
    float* __restrict__ qw)
{
    __shared__ float red[256];
    const int b = blockIdx.x / QLEN;
    const int j = blockIdx.x % QLEN;
    const int d = threadIdx.x;
    const size_t row = (size_t)blockIdx.x;
    const float v = q[row * HDIM + d];
    const unsigned short bv = f2bf(v);
    q_bf[row * HDIM + d] = bv;
    qT_bf[((size_t)b * HDIM + d) * QLEN + j] = bv;
    red[d] = v * w_q[d];
    __syncthreads();
    #pragma unroll
    for (int s = 128; s > 0; s >>= 1) {
        if (d < s) red[d] += red[d + s];
        __syncthreads();
    }
    if (d == 0) qw[row] = red[0] + b_q[0];
}

// ---------------------------------------------------------------------------
// Main fused kernel. Per wave: 16 c-rows, full flash pass over QLEN.
// Prologue reads c (f32) once, packs the c^T B-operand (scaled by w_cq) into
// registers, and computes the row-constant c.w_c + b_c + b_cq in the same pass.
// GEMM1 computes S^T = q @ (c*w_cq)^T whose C-layout equals GEMM2's A layout
// for P=exp(S^T-m) (in-lane bf16 packing, no LDS transpose). GEMM2 B-tiles
// (q^T chunks) are TDM-prefetched into padded LDS, double-buffered.
// ---------------------------------------------------------------------------
__global__ __launch_bounds__(128) void k_flash(
    const float* __restrict__ c,
    const float* __restrict__ w_c, const float* __restrict__ b_c,
    const float* __restrict__ w_cq, const float* __restrict__ b_cq,
    const unsigned short* __restrict__ q_bf,
    const unsigned short* __restrict__ qT_bf,
    const float* __restrict__ qw,
    float* __restrict__ m_out, float* __restrict__ out)
{
    // 256 d-rows x (64B data + 16B pad) per buffer = 20.5KB; x2 = 41KB.
    __shared__ __align__(16) unsigned short qtt[NQBUF][256 * 40];

    const int w    = threadIdx.x >> 5;
    const int lane = threadIdx.x & 31;
    const int lo   = lane & 15;
    const int hi   = lane >> 4;

    const int tilesPerBatch = CLEN / 64;
    const int b  = blockIdx.x / tilesPerBatch;
    const int i0 = (blockIdx.x % tilesPerBatch) * 64 + w * 16;

    const unsigned short* qbase = q_bf + ((size_t)b * QLEN) * HDIM;
    const unsigned short* qTb   = qT_bf + (size_t)b * HDIM * QLEN;
    const float* qwb = qw + (size_t)b * QLEN;

    // ---- prologue: build c^T B-operand (bf16, scaled by w_cq) + row constant.
    // Lane covers row i0+lo, K elements [k0+hi*8, +8) and [k0+16+hi*8, +8) per chunk.
    const float* crowf = c + ((size_t)(b * CLEN + i0 + lo)) * HDIM;
    ABPack bc[8];
    float cwsum = 0.0f;
    #pragma unroll
    for (int k0 = 0; k0 < HDIM; k0 += 32) {
        const float4* cf  = reinterpret_cast<const float4*>(crowf + k0 + hi * 8);
        const float4* wqf = reinterpret_cast<const float4*>(w_cq + k0 + hi * 8);
        const float4* wcf = reinterpret_cast<const float4*>(w_c + k0 + hi * 8);
        #pragma unroll
        for (int e = 0; e < 4; e++) {
            const float4 f  = cf[e + (e >> 1) * 2];
            const float4 wq = wqf[e + (e >> 1) * 2];
            const float4 wc = wcf[e + (e >> 1) * 2];
            cwsum += f.x * wc.x + f.y * wc.y + f.z * wc.z + f.w * wc.w;
            bc[k0 / 32].w[e * 2] =
                ((unsigned)f2bf(f.y * wq.y) << 16) | f2bf(f.x * wq.x);
            bc[k0 / 32].w[e * 2 + 1] =
                ((unsigned)f2bf(f.w * wq.w) << 16) | f2bf(f.z * wq.z);
        }
    }
    const float cw_lane =
        cwsum + __shfl_xor(cwsum, 16, 32) + b_c[0] + b_cq[0];

    v8f acc[16];
    #pragma unroll
    for (int t = 0; t < 16; t++)
        #pragma unroll
        for (int e = 0; e < 8; e++) acc[t][e] = 0.0f;

    float m_run = -INFINITY;
    float l_run = 0.0f;

#if TDM_PATH
    if (w == 0)
        tdm_load_qtt(qTb + 0, (unsigned)(uintptr_t)(void*)&qtt[0][0]);
#endif

    for (int n = 0; n < QLEN / 32; n++) {
        const int j0 = n * 32;

#if TDM_PATH
        if (w == 0) __builtin_amdgcn_s_wait_tensorcnt(0);  // chunk n resident
        __syncthreads();           // all waves done with buf[(n+1)&1] from n-1
        if (w == 0 && n + 1 < QLEN / 32)
            tdm_load_qtt(qTb + (j0 + 32),
                         (unsigned)(uintptr_t)(void*)&qtt[(n + 1) & 1][0]);
        const unsigned short* qt = &qtt[n & 1][0];
#else
        __syncthreads();
        {   // cooperative copy: 256 rows x 4 uint4 = 1024 uint4, 8 per thread
            const int tid = threadIdx.x;
            #pragma unroll
            for (int e = 0; e < 8; e++) {
                const int idx = e * 128 + tid;
                const int row = idx >> 2;
                const int c4  = idx & 3;
                *reinterpret_cast<uint4*>(&qtt[0][row * 40 + c4 * 8]) =
                    *reinterpret_cast<const uint4*>(qTb + (size_t)row * QLEN +
                                                    j0 + c4 * 8);
            }
        }
        __syncthreads();
        const unsigned short* qt = &qtt[0][0];
#endif

        // ---- GEMM1: S^T (2 tiles of 16x16) = q_chunk @ (c*w_cq)^T, K=HDIM ----
        v8f T0, T1;
        #pragma unroll
        for (int e = 0; e < 8; e++) { T0[e] = 0.0f; T1[e] = 0.0f; }

        const unsigned short* q0row = qbase + (size_t)(j0 + lo) * HDIM;
        const unsigned short* q1row = q0row + 16 * HDIM;

        ABPack qa0, qa1, qa0n, qa1n;
        qa0.u[0] = *reinterpret_cast<const uint4*>(q0row + hi * 8);
        qa0.u[1] = *reinterpret_cast<const uint4*>(q0row + 16 + hi * 8);
        qa1.u[0] = *reinterpret_cast<const uint4*>(q1row + hi * 8);
        qa1.u[1] = *reinterpret_cast<const uint4*>(q1row + 16 + hi * 8);

        #pragma unroll
        for (int k0 = 0; k0 < HDIM; k0 += 32) {
            if (k0 + 32 < HDIM) {  // prefetch next-k A tiles during this WMMA
                const int kn = k0 + 32;
                qa0n.u[0] = *reinterpret_cast<const uint4*>(q0row + kn + hi * 8);
                qa0n.u[1] = *reinterpret_cast<const uint4*>(q0row + kn + 16 + hi * 8);
                qa1n.u[0] = *reinterpret_cast<const uint4*>(q1row + kn + hi * 8);
                qa1n.u[1] = *reinterpret_cast<const uint4*>(q1row + kn + 16 + hi * 8);
            }
            T0 = __builtin_amdgcn_wmma_f32_16x16x32_bf16(
                false, qa0.v, false, bc[k0 / 32].v, (short)0, T0, false, false);
            T1 = __builtin_amdgcn_wmma_f32_16x16x32_bf16(
                false, qa1.v, false, bc[k0 / 32].v, (short)0, T1, false, false);
            qa0 = qa0n;
            qa1 = qa1n;
        }

        // ---- online softmax: this lane owns i = i0+lo; j element (T,r,half) ----
        float sv[16];
        #pragma unroll
        for (int r = 0; r < 8; r++) {
            sv[r]     = T0[r] + qwb[j0 + r + hi * 8] + cw_lane;
            sv[8 + r] = T1[r] + qwb[j0 + 16 + r + hi * 8] + cw_lane;
        }
        float cmax = sv[0];
        #pragma unroll
        for (int e = 1; e < 16; e++) cmax = fmaxf(cmax, sv[e]);
        cmax = fmaxf(cmax, __shfl_xor(cmax, 16, 32));   // join lane halves
        const float nm = fmaxf(m_run, cmax);
        const float sc = __expf(m_run - nm);
        float p[16], rs = 0.0f;
        #pragma unroll
        for (int e = 0; e < 16; e++) { p[e] = __expf(sv[e] - nm); rs += p[e]; }
        rs += __shfl_xor(rs, 16, 32);
        l_run = l_run * sc + rs;
        m_run = nm;

        // rescale c2q accumulator: row r of acc needs scale of i = r + hi*8
        float scr[8];
        #pragma unroll
        for (int r = 0; r < 8; r++) scr[r] = __shfl(sc, r + hi * 8, 32);
        #pragma unroll
        for (int t = 0; t < 16; t++)
            #pragma unroll
            for (int r = 0; r < 8; r++) acc[t][r] *= scr[r];

        // ---- pack P into the GEMM2 A operand: pure in-lane bf16 packing ----
        ABPack pa;
        #pragma unroll
        for (int v = 0; v < 4; v++) {
            pa.w[v]     = ((unsigned)f2bf(p[2 * v + 1]) << 16) | f2bf(p[2 * v]);
            pa.w[4 + v] = ((unsigned)f2bf(p[8 + 2 * v + 1]) << 16) | f2bf(p[8 + 2 * v]);
        }

        // ---- GEMM2: acc(16 x 256) += P(16x32) @ Q(32x256), B from padded LDS ----
        #pragma unroll
        for (int t = 0; t < 16; t++) {
            const unsigned short* qr = qt + (t * 16 + lo) * 40 + hi * 8;
            ABPack bq;
            bq.u[0] = *reinterpret_cast<const uint4*>(qr);
            bq.u[1] = *reinterpret_cast<const uint4*>(qr + 16);
            acc[t] = __builtin_amdgcn_wmma_f32_16x16x32_bf16(
                false, pa.v, false, bq.v, (short)0, acc[t], false, false);
        }
    }

    // ---- epilogue: normalize, write c2q into out cols [256,512), emit m ----
    const float inv = 1.0f / l_run;
    float invr[8];
    #pragma unroll
    for (int r = 0; r < 8; r++) invr[r] = __shfl(inv, r + hi * 8, 32);

    float* outb = out + (size_t)b * CLEN * (4 * HDIM);
    #pragma unroll
    for (int t = 0; t < 16; t++)
        #pragma unroll
        for (int r = 0; r < 8; r++)
            outb[(size_t)(i0 + r + hi * 8) * (4 * HDIM) + HDIM + t * 16 + lo] =
                acc[t][r] * invr[r];

    if (hi == 0)  // lanes 0..15 hold stats for rows i0+0..15
        m_out[(size_t)b * CLEN + i0 + lo] = m_run;
}

// ---------------------------------------------------------------------------
// q2c stage 1: per-batch max and 1/Z over m (tiny).
// ---------------------------------------------------------------------------
__global__ __launch_bounds__(256) void k_stats(
    const float* __restrict__ m, float* __restrict__ stats)
{
    __shared__ float red[256];
    const int b = blockIdx.x;
    const int t = threadIdx.x;

    float lm = -INFINITY;
    for (int i = t; i < CLEN; i += 256)
        lm = fmaxf(lm, m[(size_t)b * CLEN + i]);
    red[t] = lm;
    __syncthreads();
    #pragma unroll
    for (int s = 128; s > 0; s >>= 1) {
        if (t < s) red[t] = fmaxf(red[t], red[t + s]);
        __syncthreads();
    }
    const float M = red[0];
    __syncthreads();

    float ls = 0.0f;
    for (int i = t; i < CLEN; i += 256)
        ls += __expf(m[(size_t)b * CLEN + i] - M);
    red[t] = ls;
    __syncthreads();
    #pragma unroll
    for (int s = 128; s > 0; s >>= 1) {
        if (t < s) red[t] += red[t + s];
        __syncthreads();
    }
    if (t == 0) { stats[2 * b] = M; stats[2 * b + 1] = 1.0f / red[0]; }
}

// ---------------------------------------------------------------------------
// q2c stage 2: 64-row partial weighted sums of c (1024 blocks, deterministic).
// ---------------------------------------------------------------------------
__global__ __launch_bounds__(256) void k_q2c_part(
    const float* __restrict__ m, const float* __restrict__ c,
    const float* __restrict__ stats, float* __restrict__ part)
{
    __shared__ float coef[64];
    const int b  = blockIdx.x >> 6;
    const int g  = blockIdx.x & 63;
    const int i0 = g * 64;
    const int t  = threadIdx.x;

    const float M = stats[2 * b];
    if (t < 64)
        coef[t] = __expf(m[(size_t)b * CLEN + i0 + t] - M);
    __syncthreads();

    float acc = 0.0f;
    const float* cb = c + ((size_t)b * CLEN + i0) * HDIM;
    #pragma unroll 4
    for (int ii = 0; ii < 64; ii++)
        acc += coef[ii] * cb[(size_t)ii * HDIM + t];
    part[((size_t)b * 64 + g) * HDIM + t] = acc;
}

// ---------------------------------------------------------------------------
// q2c stage 3: reduce 64 partials per channel, apply 1/Z (deterministic).
// ---------------------------------------------------------------------------
__global__ __launch_bounds__(256) void k_q2c_red(
    const float* __restrict__ part, const float* __restrict__ stats,
    float* __restrict__ q2c)
{
    const int b = blockIdx.x;
    const int t = threadIdx.x;
    float acc = 0.0f;
    #pragma unroll 4
    for (int g = 0; g < 64; g++)
        acc += part[((size_t)b * 64 + g) * HDIM + t];
    q2c[b * HDIM + t] = acc * stats[2 * b + 1];
}

// ---------------------------------------------------------------------------
// Final assembly: out = [c, c2q, c*c2q, c*q2c]; c2q already in cols [256,512).
// ---------------------------------------------------------------------------
__global__ __launch_bounds__(256) void k_final(
    const float* __restrict__ c, const float* __restrict__ q2c,
    float* __restrict__ out)
{
    const size_t row = blockIdx.x;
    const int d = threadIdx.x;
    const int b = (int)(row / CLEN);
    const float cv = c[row * HDIM + d];
    float* o = out + row * (4 * HDIM);
    const float c2q = o[HDIM + d];
    o[d] = cv;
    o[2 * HDIM + d] = cv * c2q;
    o[3 * HDIM + d] = cv * q2c[b * HDIM + d];
}

// ---------------------------------------------------------------------------
// Workspace layout (bytes):
//   qw    @ 0       : 16*512*4      =   32768 (pad to 131072)
//   m     @ 131072  : 16*4096*4     =  262144
//   q2c   @ 393216  : 16*256*4      =   16384
//   stats @ 409600  : 16*2*4        =     128 (pad to 256)
//   part  @ 409856  : 16*64*256*4   = 1048576
//   q_bf  @ 1458432 : 16*512*256*2  = 4194304
//   qT_bf @ 5652736 : 16*512*256*2  = 4194304
//   total 9847040 bytes (~9.4 MB)
// ---------------------------------------------------------------------------
extern "C" void kernel_launch(void* const* d_in, const int* in_sizes, int n_in,
                              void* d_out, int out_size, void* d_ws, size_t ws_size,
                              hipStream_t stream)
{
    (void)in_sizes; (void)n_in; (void)out_size; (void)ws_size;

    const float* c    = (const float*)d_in[0];
    const float* q    = (const float*)d_in[1];
    const float* w_c  = (const float*)d_in[2];
    const float* b_c  = (const float*)d_in[3];
    const float* w_q  = (const float*)d_in[4];
    const float* b_q  = (const float*)d_in[5];
    const float* w_cq = (const float*)d_in[6];
    const float* b_cq = (const float*)d_in[7];
    float* out = (float*)d_out;

    char* ws = (char*)d_ws;
    float*          qw    = (float*)(ws + 0);
    float*          m     = (float*)(ws + 131072);
    float*          q2c   = (float*)(ws + 393216);
    float*          stats = (float*)(ws + 409600);
    float*          part  = (float*)(ws + 409856);
    unsigned short* q_bf  = (unsigned short*)(ws + 1458432);
    unsigned short* qT_bf = (unsigned short*)(ws + 5652736);

    k_prep_q<<<BATCH * QLEN, 256, 0, stream>>>(q, w_q, b_q, q_bf, qT_bf, qw);
    k_flash<<<BATCH * (CLEN / 64), 128, 0, stream>>>(
        c, w_c, b_c, w_cq, b_cq, q_bf, qT_bf, qw, m, out);
    k_stats<<<BATCH, 256, 0, stream>>>(m, stats);
    k_q2c_part<<<BATCH * 64, 256, 0, stream>>>(m, c, stats, part);
    k_q2c_red<<<BATCH, 256, 0, stream>>>(part, stats, q2c);
    k_final<<<BATCH * CLEN, 256, 0, stream>>>(c, q2c, out);
}
